// MultiheadChannelAttention_74612171866629
// MI455X (gfx1250) — matrix operand order repaired and checked by reading, not compiled, not run on hardware
//
#include <hip/hip_runtime.h>
#include <hip/hip_bf16.h>
#include <math.h>

// Problem constants (baked from reference: b=2, c=192, h=w=256, input_num=3)
#define HW_N   65536L
#define CCH    192
#define BATCH  2
#define HEADS  6
#define EPSF   1e-12f

// Flip to 0 if the toolchain rejects the gfx1250 async-copy mnemonics.
#define USE_ASYNC_COPY 1

typedef __attribute__((ext_vector_type(16))) __bf16 v16bf;
typedef __attribute__((ext_vector_type(8)))  __bf16 v8bf;
typedef __attribute__((ext_vector_type(8)))  float  v8f;

// Concatenate two 16-byte chunks into one 16-element bf16 fragment register set.
__device__ __forceinline__ v16bf frag_cat(const __bf16* p0, const __bf16* p1) {
  union { v16bf v; v8bf h[2]; } u;
  u.h[0] = *reinterpret_cast<const v8bf*>(p0);
  u.h[1] = *reinterpret_cast<const v8bf*>(p1);
  return u.v;
}

// gfx1250 async global->LDS 16B copy (ASYNCcnt-tracked, no VGPR round trip).
// Generic LDS pointers carry the wave-relative LDS byte offset in their low
// 32 bits (ISA 10.2 aperture rules), which is exactly what VDST wants.
__device__ __forceinline__ void async_copy_b128(const __bf16* gsrc, __bf16* ldst) {
#if USE_ASYNC_COPY
  unsigned lds_off = (unsigned)(unsigned long)ldst;
  asm volatile("global_load_async_to_lds_b128 %0, %1, off"
               :: "v"(lds_off), "v"((unsigned long)gsrc) : "memory");
#else
  *reinterpret_cast<v8bf*>(ldst) = *reinterpret_cast<const v8bf*>(gsrc);
#endif
}
__device__ __forceinline__ void async_wait0() {
#if USE_ASYNC_COPY
  asm volatile("s_wait_asynccnt 0x0" ::: "memory");
#endif
}

// ---------------------------------------------------------------------------
// x (f32) -> x_bf (bf16), vectorized 8 elements/thread.
// ---------------------------------------------------------------------------
__global__ __launch_bounds__(256)
void convert_x_kernel(const float* __restrict__ x, __bf16* __restrict__ xb) {
  const long i = ((long)blockIdx.x * 256 + threadIdx.x) * 8;
  const float4 a = *reinterpret_cast<const float4*>(x + i);
  const float4 b = *reinterpret_cast<const float4*>(x + i + 4);
  union { v8bf v; __bf16 e[8]; } u;
  u.e[0] = (__bf16)a.x; u.e[1] = (__bf16)a.y; u.e[2] = (__bf16)a.z; u.e[3] = (__bf16)a.w;
  u.e[4] = (__bf16)b.x; u.e[5] = (__bf16)b.y; u.e[6] = (__bf16)b.z; u.e[7] = (__bf16)b.w;
  *reinterpret_cast<v8bf*>(xb + i) = u.v;
}

// ---------------------------------------------------------------------------
// Weight prep: fold the 3x channel-tiling into qw (192x192 -> 192x64) and
// convert all GEMM A-operands to bf16 once.
// ---------------------------------------------------------------------------
__global__ void prep_weights_kernel(const float* __restrict__ qw,
                                    const float* __restrict__ kw,
                                    const float* __restrict__ vw,
                                    __bf16* __restrict__ qwE,
                                    __bf16* __restrict__ kwB,
                                    __bf16* __restrict__ vwB) {
  int idx = blockIdx.x * 256 + threadIdx.x;
  if (idx >= CCH * CCH) return;
  kwB[idx] = (__bf16)kw[idx];
  vwB[idx] = (__bf16)vw[idx];
  int o = idx / CCH, c = idx % CCH;
  if (c < 64)
    qwE[o * 64 + c] =
        (__bf16)(qw[o * CCH + c] + qw[o * CCH + 64 + c] + qw[o * CCH + 128 + c]);
}

// ---------------------------------------------------------------------------
// BF16 WMMA GEMM:  Out[b] = A[b] (MxK, bf16) @ B[b] (KxN, bf16) + bias
// Block tile: 64(M) x 128(N), 256 threads = 8 waves, each wave owns a
// 16(M) x 64(N) strip = 4 accumulators of v_wmma_f32_16x16x32_bf16.
// A panel + double-buffered B tiles are staged with async global->LDS
// copies; K is a template parameter so all staging index math is constant.
// ---------------------------------------------------------------------------
template <int KT_, bool OUT_BF16, bool NORMSQ>
__global__ __launch_bounds__(256)
void gemm_wmma_kernel(const __bf16* __restrict__ A, long aStride,
                      const __bf16* __restrict__ B, long bStride,
                      const float* __restrict__ bias,
                      void* __restrict__ OutV,
                      float* __restrict__ normsq,
                      int M, long N) {
  const int  bb   = blockIdx.z;
  const int  mBlk = blockIdx.y * 64;
  const long nBlk = (long)blockIdx.x * 128;
  A += (long)bb * aStride;
  B += (long)bb * bStride + nBlk;

  __shared__ __bf16 sA[64 * KT_];        // full A panel for this M block
  __shared__ __bf16 sB[2][32 * 128];     // double-buffered K-step of B

  const int tid  = threadIdx.x;
  const int wave = tid >> 5;
  const int lane = tid & 31;
  const int mt   = wave >> 1;            // 0..3 : which 16-row M tile
  const int ng   = wave & 1;             // 0..1 : which 64-col N group

  // Stage A panel (async): KT_/8 16B chunks per row, 64 rows.
  constexpr int CPR = KT_ / 8;
  for (int c = tid; c < 64 * CPR; c += 256) {
    const int m = c / CPR, kc = (c % CPR) * 8;   // CPR is compile-time
    async_copy_b128(&A[(long)(mBlk + m) * KT_ + kc], &sA[m * KT_ + kc]);
  }
  // Stage B tile s into buffer (s&1): 512 16B chunks.
  auto stage_b = [&](int s) {
    const __bf16* src = B + (long)s * 32 * N;
    __bf16* dst = sB[s & 1];
    for (int c = tid; c < 512; c += 256) {
      const int row = c >> 4, col = (c & 15) * 8;
      async_copy_b128(&src[(long)row * N + col], &dst[row * 128 + col]);
    }
  };
  constexpr int NSTEPS = KT_ / 32;
  stage_b(0);

  v8f acc[4] = {};
  const int am = mt * 16 + (lane & 15);
  const int hh = lane >> 4;

  for (int s = 0; s < NSTEPS; ++s) {
    async_wait0();        // this wave's async copies for tile s (and A) landed
    __syncthreads();      // ... and everyone else's
    if (s + 1 < NSTEPS) stage_b(s + 1);     // overlap next tile with math
    if (s + 2 < NSTEPS)                     // L2 warm for the tile after next
      __builtin_prefetch(B + (long)(s + 2) * 32 * N + (long)(tid >> 3) * N + (tid & 7) * 16, 0, 1);

    const int k0 = s * 32;
    // A fragment: lane = M row (two halves), K pattern {h*8..+7, 16+h*8..+7}
    const __bf16* ap = &sA[am * KT_ + k0 + hh * 8];
    const v16bf afrag = frag_cat(ap, ap + 16);
    const __bf16* bbase = &sB[s & 1][hh * 8 * 128 + ng * 64 + (lane & 15)];
    for (int j = 0; j < 4; ++j) {
      // B fragment: lane = N column (two halves), K pattern {h*8..+7, 16+h*8..+7}
      v16bf bfrag;
      for (int e = 0; e < 16; ++e) bfrag[e] = bbase[j * 16 + ((e & 7) + (e >> 3) * 16) * 128];
      acc[j] = __builtin_amdgcn_wmma_f32_16x16x32_bf16(
          false, afrag, false, bfrag, (short)0, acc[j], false, false);
    }
  }

  // Epilogue: C layout — lanes 0-15: VGPR r -> M=r; lanes 16-31: M=r+8.
  const int  rbase = (lane < 16) ? 0 : 8;
  const long nn0   = nBlk + ng * 64 + (lane & 15);
  for (int r = 0; r < 8; ++r) {
    const int   m  = mBlk + mt * 16 + rbase + r;
    const float bv = bias[m];
    float sq = 0.f;
    for (int j = 0; j < 4; ++j) {
      float v = acc[j][r] + bv;
      long idx = (long)bb * M * N + (long)m * N + nn0 + j * 16;
      if (OUT_BF16) ((__bf16*)OutV)[idx] = (__bf16)v;
      else          ((float*)OutV)[idx]  = v;
      sq += v * v;
    }
    if (NORMSQ) {
      for (int off = 1; off < 16; off <<= 1) sq += __shfl_xor(sq, off, 32);
      if ((lane & 15) == 0) unsafeAtomicAdd(&normsq[bb * M + m], sq);
    }
  }
}

// ---------------------------------------------------------------------------
// 32x32 LDS tile transpose of K (b,192,N) -> KT (b,N,192), bf16.
// ---------------------------------------------------------------------------
__global__ __launch_bounds__(256)
void transpose_bf16_kernel(const __bf16* __restrict__ src,
                           __bf16* __restrict__ dst, long N) {
  __shared__ __bf16 tile[32][33];
  const long n0 = (long)blockIdx.x * 32;
  const int  c0 = blockIdx.y * 32;
  const long bS = (long)CCH * N;
  const __bf16* s = src + blockIdx.z * bS;
  __bf16*       d = dst + blockIdx.z * bS;
  const int tx = threadIdx.x & 31, ty = threadIdx.x >> 5;
  for (int i = 0; i < 32; i += 8)
    tile[ty + i][tx] = s[(long)(c0 + ty + i) * N + n0 + tx];
  __syncthreads();
  for (int i = 0; i < 32; i += 8)
    d[(n0 + ty + i) * CCH + c0 + tx] = tile[tx][ty + i];
}

// ---------------------------------------------------------------------------
// attn_raw[b,h] += Q_h (32xN) @ K_h^T (Nx32), split-K over grid.y.
// A-frags come straight from row-major Q; B-frags from pre-transposed KT,
// A-frag loads are contiguous 16B; B-frag elements are 2-byte loads CCH apart. Per-block LDS
// f32 reduction, then one global atomic per element per block.
// ---------------------------------------------------------------------------
__global__ __launch_bounds__(256)
void attn_qk_kernel(const __bf16* __restrict__ Q, const __bf16* __restrict__ KT,
                    float* __restrict__ raw, long N, int nPerBlock) {
  const int head = blockIdx.x % HEADS;
  const int bb   = blockIdx.x / HEADS;
  const __bf16* Qh  = Q  + ((long)bb * CCH + head * 32) * N;
  const __bf16* KTb = KT + (long)bb * N * CCH;
  const int wave = threadIdx.x >> 5, lane = threadIdx.x & 31;

  __shared__ float sAcc[32 * 32];
  for (int i = threadIdx.x; i < 1024; i += 256) sAcc[i] = 0.f;
  __syncthreads();

  v8f acc[2][2] = {};
  const int  hh = lane >> 4, ml = lane & 15;
  const long nEnd = (long)(blockIdx.y + 1) * nPerBlock;
  for (long n0 = (long)blockIdx.y * nPerBlock + wave * 32; n0 < nEnd; n0 += 256) {
    v16bf a[2], b[2];
    for (int t = 0; t < 2; ++t) {
      const __bf16* ap = Qh + (long)(t * 16 + ml) * N + n0 + hh * 8;
      a[t] = frag_cat(ap, ap + 16);
      const __bf16* bp = KTb + (n0 + hh * 8) * CCH + head * 32 + t * 16 + ml;
      for (int e = 0; e < 16; ++e) b[t][e] = bp[(long)((e & 7) + (e >> 3) * 16) * CCH];
    }
    for (int mi = 0; mi < 2; ++mi)
      for (int ni = 0; ni < 2; ++ni)
        acc[mi][ni] = __builtin_amdgcn_wmma_f32_16x16x32_bf16(
            false, a[mi], false, b[ni], (short)0, acc[mi][ni], false, false);
  }

  const int rbase = (lane < 16) ? 0 : 8, nl = lane & 31 & 15;
  for (int mi = 0; mi < 2; ++mi)
    for (int ni = 0; ni < 2; ++ni)
      for (int r = 0; r < 8; ++r)
        atomicAdd(&sAcc[(mi * 16 + rbase + r) * 32 + ni * 16 + nl], acc[mi][ni][r]);
  __syncthreads();
  float* out = raw + (long)blockIdx.x * 1024;
  for (int i = threadIdx.x; i < 1024; i += 256)
    unsafeAtomicAdd(&out[i], sAcc[i]);
}

// ---------------------------------------------------------------------------
// Per-head softmax with the deferred q/k L2 normalization folded in.
// One wave32 per (batch,head); lane = row.
// ---------------------------------------------------------------------------
__global__ void softmax_kernel(const float* __restrict__ raw,
                               const float* __restrict__ qns,
                               const float* __restrict__ kns,
                               float* __restrict__ attn) {
  const int head = blockIdx.x % HEADS, bb = blockIdx.x / HEADS;
  const int lane = threadIdx.x;  // row c' in [0,32)
  const float* r = raw + (long)blockIdx.x * 1024 + lane * 32;
  const float qn = fmaxf(sqrtf(qns[bb * CCH + head * 32 + lane]), EPSF);
  float vals[32], mx = -INFINITY;
  for (int d = 0; d < 32; ++d) {
    const float kn = fmaxf(sqrtf(kns[bb * CCH + head * 32 + d]), EPSF);
    vals[d] = r[d] / (qn * kn);
    mx = fmaxf(mx, vals[d]);
  }
  float s = 0.f;
  for (int d = 0; d < 32; ++d) { vals[d] = __expf(vals[d] - mx); s += vals[d]; }
  const float inv = 1.f / s;
  float* o = attn + (long)blockIdx.x * 1024 + lane * 32;
  for (int d = 0; d < 32; ++d) o[d] = vals[d] * inv;
}

// ---------------------------------------------------------------------------
// W2[b] = pw @ blockdiag(attn[b])  (192x192 per batch, bf16 output).
// Fuses attn@v with the final 1x1 conv into a single big GEMM.
// ---------------------------------------------------------------------------
__global__ void build_w2_kernel(const float* __restrict__ pw,
                                const float* __restrict__ attn,
                                __bf16* __restrict__ W2) {
  const int idx = blockIdx.x * 256 + threadIdx.x;
  if (idx >= BATCH * CCH * CCH) return;
  const int kk = idx % CCH;
  const int o  = (idx / CCH) % CCH;
  const int bb = idx / (CCH * CCH);
  const int h = kk >> 5, d = kk & 31;
  const float* a = attn + ((long)bb * HEADS + h) * 1024;  // [c'][d]
  const float* p = pw + o * CCH + h * 32;
  float s = 0.f;
  for (int c = 0; c < 32; ++c) s += p[c] * a[c * 32 + d];
  W2[idx] = (__bf16)s;
}

// ---------------------------------------------------------------------------
// Host orchestration. Workspace (~202MB): x_bf, Q, K, V bf16 planes (KT
// aliases x_bf, which is dead after the V GEMM) + small buffers.
// ---------------------------------------------------------------------------
extern "C" void kernel_launch(void* const* d_in, const int* in_sizes, int n_in,
                              void* d_out, int out_size, void* d_ws, size_t ws_size,
                              hipStream_t stream) {
  (void)in_sizes; (void)n_in; (void)out_size; (void)ws_size;
  const float* x  = (const float*)d_in[0];
  const float* qw = (const float*)d_in[1];
  const float* qb = (const float*)d_in[2];
  const float* kw = (const float*)d_in[3];
  const float* kb = (const float*)d_in[4];
  const float* vw = (const float*)d_in[5];
  const float* vb = (const float*)d_in[6];
  const float* pw = (const float*)d_in[7];
  const float* pb = (const float*)d_in[8];
  // d_in[9] = input_num (always 3; folded into qw_eff at prep time)

  const long N = HW_N;
  char* w = (char*)d_ws;
  auto take = [&](size_t bytes) {
    void* p = (void*)w;
    w += (bytes + 255) & ~(size_t)255;
    return p;
  };
  const size_t planeE = (size_t)BATCH * CCH * N;        // elements per plane
  __bf16* xb   = (__bf16*)take(planeE * 2);
  __bf16* Q    = (__bf16*)take(planeE * 2);
  __bf16* Kb   = (__bf16*)take(planeE * 2);
  __bf16* V    = (__bf16*)take(planeE * 2);
  __bf16* KT   = xb;  // alias: xb is dead once the V GEMM has run
  __bf16* qwE  = (__bf16*)take((size_t)CCH * 64 * 2);
  __bf16* kwB  = (__bf16*)take((size_t)CCH * CCH * 2);
  __bf16* vwB  = (__bf16*)take((size_t)CCH * CCH * 2);
  __bf16* W2   = (__bf16*)take((size_t)BATCH * CCH * CCH * 2);
  float*  qns  = (float*)take((size_t)BATCH * CCH * 4);
  float*  kns  = (float*)take((size_t)BATCH * CCH * 4);
  float*  raw  = (float*)take((size_t)BATCH * HEADS * 1024 * 4);
  float*  attn = (float*)take((size_t)BATCH * HEADS * 1024 * 4);

  // Atomic accumulators must be zeroed every call (graph-replay safe).
  hipMemsetAsync(qns, 0, (size_t)BATCH * CCH * 4, stream);
  hipMemsetAsync(kns, 0, (size_t)BATCH * CCH * 4, stream);
  hipMemsetAsync(raw, 0, (size_t)BATCH * HEADS * 1024 * 4, stream);

  convert_x_kernel<<<(unsigned)(planeE / 8 / 256), 256, 0, stream>>>(x, xb);
  prep_weights_kernel<<<(CCH * CCH + 255) / 256, 256, 0, stream>>>(
      qw, kw, vw, qwE, kwB, vwB);

  const dim3 gG(N / 128, CCH / 64, BATCH);
  // Q = qw_eff(192x64) @ x[:,64:128,:] + qb   (with sum-of-squares)
  gemm_wmma_kernel<64, true, true><<<gG, 256, 0, stream>>>(
      qwE, 0, xb + 64 * N, (long)CCH * N, qb, Q, qns, CCH, N);
  // K = kw @ x + kb   (with sum-of-squares)
  gemm_wmma_kernel<192, true, true><<<gG, 256, 0, stream>>>(
      kwB, 0, xb, (long)CCH * N, kb, Kb, kns, CCH, N);
  // V = vw @ x + vb
  gemm_wmma_kernel<192, true, false><<<gG, 256, 0, stream>>>(
      vwB, 0, xb, (long)CCH * N, vb, V, nullptr, CCH, N);

  transpose_bf16_kernel<<<dim3(N / 32, CCH / 32, BATCH), 256, 0, stream>>>(Kb, KT, N);

  attn_qk_kernel<<<dim3(BATCH * HEADS, 16), 256, 0, stream>>>(
      Q, KT, raw, N, (int)(N / 16));

  softmax_kernel<<<BATCH * HEADS, 32, 0, stream>>>(raw, qns, kns, attn);

  build_w2_kernel<<<(BATCH * CCH * CCH + 255) / 256, 256, 0, stream>>>(pw, attn, W2);

  // final = W2[b] @ V[b] + pb  -> f32 output
  gemm_wmma_kernel<192, false, false><<<gG, 256, 0, stream>>>(
      W2, (long)CCH * CCH, V, (long)CCH * N, pb, d_out, nullptr, CCH, N);
}
